// MultiViewDGT_22144851378799
// MI455X (gfx1250) — compile-verified, matrix operand order checked by reading, not run on hardware
//
#include <hip/hip_runtime.h>
#include <hip/hip_bf16.h>

// ---------------- problem constants (match reference) ----------------
#define NN   10000   // nodes
#define XDIM 64
#define DCH  128     // hidden
#define NH   4       // heads
#define HD   512     // heads*D
#define NVW  4       // views
#define EVN  20000   // edges per view
#define GN   1000    // portfolios
#define LLN  100000  // portfolio lines

typedef __attribute__((ext_vector_type(16))) _Float16 v16h;
typedef __attribute__((ext_vector_type(8)))  _Float16 v8h;
typedef __attribute__((ext_vector_type(8)))  float    v8f;

#define CAT16(lo, hi) __builtin_shufflevector(lo, hi, 0,1,2,3,4,5,6,7,8,9,10,11,12,13,14,15)

// ---------------- wave32 helpers ----------------
__device__ __forceinline__ float wsum(float v) {
#pragma unroll
  for (int m = 16; m; m >>= 1) v += __shfl_xor(v, m, 32);
  return v;
}
__device__ __forceinline__ float sigm(float x) { return 1.f / (1.f + __expf(-x)); }
__device__ __forceinline__ unsigned f2o(float f) {
  unsigned u = __float_as_uint(f);
  return (u & 0x80000000u) ? ~u : (u | 0x80000000u);
}
__device__ __forceinline__ float o2f(unsigned u) {
  return __uint_as_float((u & 0x80000000u) ? (u & 0x7fffffffu) : ~u);
}

// ---------------- conversion kernels (one-time per launch) ----------------
// A (rows x K, f32 row-major) -> out (rows x Kp, f16, zero-padded K)
__global__ void cvt_pad_kernel(const float* __restrict__ A, _Float16* __restrict__ out,
                               int rows, int K, int Kp) {
  int i = blockIdx.x * blockDim.x + threadIdx.x;
  if (i >= rows * Kp) return;
  int r = i / Kp, k = i - r * Kp;
  out[i] = (k < K) ? (_Float16)A[(size_t)r * K + k] : (_Float16)0.f;
}
// B (K x N, f32 row-major) -> BT (N x Kp, f16 "column-major", zero-padded K)
__global__ void cvt_transpose_kernel(const float* __restrict__ B, _Float16* __restrict__ BT,
                                     int K, int N, int Kp) {
  int i = blockIdx.x * blockDim.x + threadIdx.x;
  if (i >= N * Kp) return;
  int n = i / Kp, k = i - n * Kp;
  BT[i] = (k < K) ? (_Float16)B[(size_t)k * N + n] : (_Float16)0.f;
}

// ---------------- WMMA GEMM: C = act(A@B + bias), 32x32 tile per wave ----------------
// A: M x Kp f16 row-major (K padded to /32). BT: N x Kp f16 (B transposed).
// C: M x N f32; optional C16: M x N f16 (feeds next GEMM). N must be /32.
// No divergent branches before the WMMAs -> EXEC all-ones as the ISA requires.
__global__ void __launch_bounds__(256)
gemm_wmma_kernel(const _Float16* __restrict__ A, const _Float16* __restrict__ BT,
                 const float* __restrict__ bias, float* __restrict__ C,
                 _Float16* __restrict__ C16, int M, int N, int Kp, int relu)
{
  const int lane = threadIdx.x & 31;
  const int wave = blockIdx.x * (blockDim.x >> 5) + (threadIdx.x >> 5);
  const int tnc  = N >> 5;
  const int tm   = wave / tnc;
  const int tn   = wave - tm * tnc;
  if (tm * 32 >= M) return;

  const int l15  = lane & 15;
  const int koff = (lane >> 4) * 8;          // hi half-wave: K+8 / K+24
  int ar0 = tm * 32 + l15;      if (ar0 >= M) ar0 = M - 1;  // clamp: safe load, store guarded
  int ar1 = tm * 32 + 16 + l15; if (ar1 >= M) ar1 = M - 1;
  const int bc0 = tn * 32 + l15;
  const int bc1 = bc0 + 16;

  const _Float16* a0p = A  + (size_t)ar0 * Kp + koff;
  const _Float16* a1p = A  + (size_t)ar1 * Kp + koff;
  const _Float16* b0p = BT + (size_t)bc0 * Kp + koff;
  const _Float16* b1p = BT + (size_t)bc1 * Kp + koff;

  v8f acc00 = {}, acc01 = {}, acc10 = {}, acc11 = {};
  for (int k = 0; k < Kp; k += 32) {
    const v16h a0 = CAT16(*(const v8h*)(a0p + k), *(const v8h*)(a0p + k + 16));
    const v16h a1 = CAT16(*(const v8h*)(a1p + k), *(const v8h*)(a1p + k + 16));
    const v16h b0 = CAT16(*(const v8h*)(b0p + k), *(const v8h*)(b0p + k + 16));
    const v16h b1 = CAT16(*(const v8h*)(b1p + k), *(const v8h*)(b1p + k + 16));
    acc00 = __builtin_amdgcn_wmma_f32_16x16x32_f16(false, a0, false, b0, (short)0, acc00, false, false);
    acc01 = __builtin_amdgcn_wmma_f32_16x16x32_f16(false, a0, false, b1, (short)0, acc01, false, false);
    acc10 = __builtin_amdgcn_wmma_f32_16x16x32_f16(false, a1, false, b0, (short)0, acc10, false, false);
    acc11 = __builtin_amdgcn_wmma_f32_16x16x32_f16(false, a1, false, b1, (short)0, acc11, false, false);
  }
  const float bs0 = bias ? bias[bc0] : 0.f;
  const float bs1 = bias ? bias[bc1] : 0.f;
  const int rbase = tm * 32 + ((lane >> 4) * 8);   // C/D layout: VGPR r -> row r (+8 hi lanes)
#pragma unroll
  for (int r = 0; r < 8; ++r) {
    const int r0 = rbase + r;        // rows 0..15 of tile
    const int r1 = rbase + 16 + r;   // rows 16..31
    if (r0 < M) {
      float v0 = acc00[r] + bs0, v1 = acc01[r] + bs1;
      if (relu) { v0 = fmaxf(v0, 0.f); v1 = fmaxf(v1, 0.f); }
      C[(size_t)r0 * N + bc0] = v0;  C[(size_t)r0 * N + bc1] = v1;
      if (C16) { C16[(size_t)r0 * N + bc0] = (_Float16)v0; C16[(size_t)r0 * N + bc1] = (_Float16)v1; }
    }
    if (r1 < M) {
      float v0 = acc10[r] + bs0, v1 = acc11[r] + bs1;
      if (relu) { v0 = fmaxf(v0, 0.f); v1 = fmaxf(v1, 0.f); }
      C[(size_t)r1 * N + bc0] = v0;  C[(size_t)r1 * N + bc1] = v1;
      if (C16) { C16[(size_t)r1 * N + bc0] = (_Float16)v0; C16[(size_t)r1 * N + bc1] = (_Float16)v1; }
    }
  }
}

// ---------------- utility kernels ----------------
__global__ void fill_kernel(float* p, float v, int n) {
  int i = blockIdx.x * blockDim.x + threadIdx.x;
  if (i < n) p[i] = v;
}

// out = LN(x (+ add)) * g + b ; wave per row, D=128; optional f16 mirror
__global__ void ln_kernel(const float* __restrict__ x, const float* __restrict__ add,
                          const float* __restrict__ g, const float* __restrict__ b,
                          float* __restrict__ out, _Float16* __restrict__ out16, int rows)
{
  const int n = blockIdx.x * (blockDim.x >> 5) + (threadIdx.x >> 5);
  const int lane = threadIdx.x & 31;
  if (n >= rows) return;
  float v[4];
#pragma unroll
  for (int j = 0; j < 4; ++j) {
    const int d = lane + 32 * j;
    v[j] = x[(size_t)n * DCH + d] + (add ? add[(size_t)n * DCH + d] : 0.f);
  }
  const float mean = wsum(v[0] + v[1] + v[2] + v[3]) * (1.f / DCH);
  float q = 0.f;
#pragma unroll
  for (int j = 0; j < 4; ++j) { const float t = v[j] - mean; q += t * t; }
  const float inv = rsqrtf(wsum(q) * (1.f / DCH) + 1e-5f);
#pragma unroll
  for (int j = 0; j < 4; ++j) {
    const int d = lane + 32 * j;
    const float o = (v[j] - mean) * inv * g[d] + b[d];
    out[(size_t)n * DCH + d] = o;
    if (out16) out16[(size_t)n * DCH + d] = (_Float16)o;
  }
}

// ---------------- edge attention (per layer/view) ----------------
__global__ void edge_alpha_kernel(const float* __restrict__ Q, const float* __restrict__ Km,
                                  const float* __restrict__ ew, const float* __restrict__ We,
                                  const int* __restrict__ src, const int* __restrict__ dst,
                                  float* __restrict__ alpha, unsigned* __restrict__ amax)
{
  const int item = blockIdx.x * (blockDim.x >> 5) + (threadIdx.x >> 5);
  const int lane = threadIdx.x & 31;
  if (item >= EVN * NH) return;
  const int e = item >> 2, hh = item & 3;
  const int s = src[e], dn = dst[e];
  const float w = ew[e];
  float part = 0.f;
#pragma unroll
  for (int j = 0; j < 4; ++j) {
    const int d = hh * DCH + lane + 32 * j;
    part += Q[(size_t)dn * HD + d] * (Km[(size_t)s * HD + d] + w * We[d]);
  }
  const float a = wsum(part) * 0.08838834764831845f;  // 1/sqrt(128)
  if (lane == 0) {
    alpha[item] = a;
    atomicMax(&amax[dn * NH + hh], f2o(a));
  }
}

__global__ void edge_exp_kernel(const float* __restrict__ alpha, const unsigned* __restrict__ amax,
                                const int* __restrict__ dst,
                                float* __restrict__ ae, float* __restrict__ asum)
{
  const int item = blockIdx.x * blockDim.x + threadIdx.x;
  if (item >= EVN * NH) return;
  const int e = item >> 2, hh = item & 3;
  const int dn = dst[e];
  const float v = __expf(alpha[item] - o2f(amax[dn * NH + hh]));
  ae[item] = v;
  atomicAdd(&asum[dn * NH + hh], v);
}

__global__ void edge_agg_kernel(const float* __restrict__ V, const float* __restrict__ ew,
                                const float* __restrict__ We,
                                const int* __restrict__ src, const int* __restrict__ dst,
                                const float* __restrict__ ae, const float* __restrict__ asum,
                                float* __restrict__ Out)
{
  const int item = blockIdx.x * (blockDim.x >> 5) + (threadIdx.x >> 5);
  const int lane = threadIdx.x & 31;
  if (item >= EVN * NH) return;
  const int e = item >> 2, hh = item & 3;
  const int s = src[e], dn = dst[e];
  const float w = ew[e];
  const float coeff = ae[item] / (asum[dn * NH + hh] + 1e-16f);
#pragma unroll
  for (int j = 0; j < 4; ++j) {
    const int d = hh * DCH + lane + 32 * j;
    atomicAdd(&Out[(size_t)dn * HD + d], (V[(size_t)s * HD + d] + w * We[d]) * coeff);
  }
}

__global__ void combine_kernel(const float* __restrict__ Out, const float* __restrict__ skip,
                               const float* __restrict__ Wb, const float* __restrict__ gvec,
                               int vi, const float* __restrict__ corr,
                               float* __restrict__ fused)
{
  const int n = blockIdx.x * (blockDim.x >> 5) + (threadIdx.x >> 5);
  const int lane = threadIdx.x & 31;
  if (n >= NN) return;
  float om[4], xr[4], part = 0.f;
#pragma unroll
  for (int j = 0; j < 4; ++j) {
    const int d = lane + 32 * j;
    const size_t base = (size_t)n * HD + d;
    const float o = 0.25f * (Out[base] + Out[base + 128] + Out[base + 256] + Out[base + 384]);
    const float x = skip[(size_t)n * DCH + d];
    om[j] = o; xr[j] = x;
    part += o * Wb[d] + x * Wb[DCH + d] + (o - x) * Wb[2 * DCH + d];
  }
  const float beta = sigm(wsum(part));
  float scale = sigm(gvec[vi]);
  if (corr) scale *= sigm(corr[0]);
#pragma unroll
  for (int j = 0; j < 4; ++j) {
    const int d = lane + 32 * j;
    fused[(size_t)n * DCH + d] += scale * (beta * xr[j] + (1.f - beta) * om[j]);
  }
}

// ---------------- portfolio kernels ----------------
__global__ void pf_accum_kernel(const float* __restrict__ h, const int* __restrict__ nodes,
                                const float* __restrict__ w, const int* __restrict__ gid,
                                float* __restrict__ Pab, float* __restrict__ Psg,
                                float* __restrict__ denom)
{
  const int l = blockIdx.x * (blockDim.x >> 5) + (threadIdx.x >> 5);
  const int lane = threadIdx.x & 31;
  if (l >= LLN) return;
  const int nd = nodes[l], g = gid[l];
  const float ww = w[l], wa = fabsf(ww);
#pragma unroll
  for (int j = 0; j < 4; ++j) {
    const int d = lane + 32 * j;
    const float hv = h[(size_t)nd * DCH + d];
    atomicAdd(&Pab[(size_t)g * DCH + d], wa * hv);
    atomicAdd(&Psg[(size_t)g * DCH + d], ww * hv);
  }
  if (lane == 0) atomicAdd(&denom[nd], wa);
}

__global__ void pf_scatter_kernel(const float* __restrict__ h, const int* __restrict__ nodes,
                                  const float* __restrict__ w, const int* __restrict__ gid,
                                  const float* __restrict__ Pab, const float* __restrict__ Psg,
                                  float* __restrict__ Vab, float* __restrict__ Vsg)
{
  const int l = blockIdx.x * (blockDim.x >> 5) + (threadIdx.x >> 5);
  const int lane = threadIdx.x & 31;
  if (l >= LLN) return;
  const int nd = nodes[l], g = gid[l];
  const float ww = w[l], wa = fabsf(ww);
#pragma unroll
  for (int j = 0; j < 4; ++j) {
    const int d = lane + 32 * j;
    const float hv = h[(size_t)nd * DCH + d];
    atomicAdd(&Vab[(size_t)nd * DCH + d], wa * (Pab[(size_t)g * DCH + d] - wa * hv));
    atomicAdd(&Vsg[(size_t)nd * DCH + d], wa * (Psg[(size_t)g * DCH + d] - ww * hv));
  }
}

// normalize by denom, L2-normalize, emit pfcat16 = [V_abs, V_sgn] (N x 256, f16)
__global__ void pf_finish_kernel(const float* __restrict__ Vab, const float* __restrict__ Vsg,
                                 const float* __restrict__ denom, _Float16* __restrict__ pfcat16)
{
  const int n = blockIdx.x * (blockDim.x >> 5) + (threadIdx.x >> 5);
  const int lane = threadIdx.x & 31;
  if (n >= NN) return;
  const float inv = 1.f / fmaxf(denom[n], 1e-8f);
  float va[4], vs[4], qa = 0.f, qs = 0.f;
#pragma unroll
  for (int j = 0; j < 4; ++j) {
    const int d = lane + 32 * j;
    va[j] = Vab[(size_t)n * DCH + d] * inv;
    vs[j] = Vsg[(size_t)n * DCH + d] * inv;
    qa += va[j] * va[j]; qs += vs[j] * vs[j];
  }
  const float na = fmaxf(sqrtf(wsum(qa)), 1e-6f);
  const float ns = fmaxf(sqrtf(wsum(qs)), 1e-6f);
#pragma unroll
  for (int j = 0; j < 4; ++j) {
    const int d = lane + 32 * j;
    pfcat16[(size_t)n * 256 + d]       = (_Float16)(va[j] / na);
    pfcat16[(size_t)n * 256 + 128 + d] = (_Float16)(vs[j] / ns);
  }
}

__global__ void add_gate_kernel(float* __restrict__ h, const float* __restrict__ pf,
                                const float* __restrict__ gate, int n)
{
  const int i = blockIdx.x * blockDim.x + threadIdx.x;
  if (i < n) h[i] += sigm(gate[0]) * pf[i];
}

// z16 = f16 concat [h | m | t], N x 384
__global__ void concat3_kernel(const float* __restrict__ a, const float* __restrict__ b,
                               const float* __restrict__ c, _Float16* __restrict__ z16)
{
  const int i = blockIdx.x * blockDim.x + threadIdx.x;
  if (i >= NN * 3 * DCH) return;
  const int n = i / (3 * DCH), col = i - n * 3 * DCH;
  float v;
  if (col < DCH)          v = a[(size_t)n * DCH + col];
  else if (col < 2 * DCH) v = b[(size_t)n * DCH + col - DCH];
  else                    v = c[(size_t)n * DCH + col - 2 * DCH];
  z16[i] = (_Float16)v;
}

__global__ void head_gemv_kernel(const float* __restrict__ hh, const float* __restrict__ W2,
                                 const float* __restrict__ b2, float* __restrict__ out)
{
  const int n = blockIdx.x * (blockDim.x >> 5) + (threadIdx.x >> 5);
  const int lane = threadIdx.x & 31;
  if (n >= NN) return;
  float part = 0.f;
#pragma unroll
  for (int j = 0; j < 4; ++j) {
    const int d = lane + 32 * j;
    part += hh[(size_t)n * DCH + d] * W2[d];
  }
  const float s = wsum(part);
  if (lane == 0) out[n] = s + b2[0];
}

// ---------------- host orchestration ----------------
extern "C" void kernel_launch(void* const* d_in, const int* in_sizes, int n_in,
                              void* d_out, int out_size, void* d_ws, size_t ws_size,
                              hipStream_t stream) {
  (void)in_sizes; (void)n_in; (void)out_size; (void)ws_size;
  // data inputs (setup_inputs dict order)
  const float* x     = (const float*)d_in[0];
  const float* mkt   = (const float*)d_in[1];
  const float* trade = (const float*)d_in[2];
  const int*   src   = (const int*)d_in[3];
  const int*   dst   = (const int*)d_in[4];
  const float* ew    = (const float*)d_in[5];
  const int*   pnod  = (const int*)d_in[6];
  const float* pw    = (const float*)d_in[7];
  const int*   pgid  = (const int*)d_in[8];
  // params dict flattened as pytree -> ASCII-sorted keys, indices 9..46
  const float* Wbeta = (const float*)d_in[9];
  const float* We    = (const float*)d_in[10];
  const float* Wk    = (const float*)d_in[11];
  const float* Wq    = (const float*)d_in[12];
  const float* Wskip = (const float*)d_in[13];
  const float* Wv    = (const float*)d_in[14];
  const float* bk    = (const float*)d_in[15];
  const float* bq    = (const float*)d_in[16];
  const float* bskip = (const float*)d_in[17];
  const float* bv    = (const float*)d_in[18];
  const float* corrg = (const float*)d_in[19];
  const float* eW1   = (const float*)d_in[20];
  const float* eW2   = (const float*)d_in[21];
  const float* eb1   = (const float*)d_in[22];
  const float* eb2   = (const float*)d_in[23];
  const float* g1    = (const float*)d_in[24];
  const float* g2    = (const float*)d_in[25];
  const float* hW1   = (const float*)d_in[26];
  const float* hW2   = (const float*)d_in[27];
  const float* hb1   = (const float*)d_in[28];
  const float* hb2   = (const float*)d_in[29];
  const float* ln0b  = (const float*)d_in[30];
  const float* ln0g  = (const float*)d_in[31];
  const float* ln1b  = (const float*)d_in[32];
  const float* ln1g  = (const float*)d_in[33];
  const float* ln2b  = (const float*)d_in[34];
  const float* ln2g  = (const float*)d_in[35];
  const float* mW1   = (const float*)d_in[36];
  const float* mW2   = (const float*)d_in[37];
  const float* mb1   = (const float*)d_in[38];
  const float* mb2   = (const float*)d_in[39];
  const float* pfW   = (const float*)d_in[40];
  const float* pfb   = (const float*)d_in[41];
  const float* pfg   = (const float*)d_in[42];
  const float* tW1   = (const float*)d_in[43];
  const float* tW2   = (const float*)d_in[44];
  const float* tb1   = (const float*)d_in[45];
  const float* tb2   = (const float*)d_in[46];

  // -------- workspace layout --------
  float* WS = (float*)d_ws;
  size_t off = 0;
  auto alloc = [&](size_t n) { float* p = WS + off; off += n; return p; };
  float* h     = alloc((size_t)NN * DCH);
  float* fused = alloc((size_t)NN * DCH);
  float* skipb = alloc((size_t)NN * DCH);
  float* bufQ  = alloc((size_t)NN * HD);
  float* bufK  = alloc((size_t)NN * HD);
  float* bufV  = alloc((size_t)NN * HD);
  float* bufO  = alloc((size_t)NN * HD);
  float* alphb = alloc((size_t)EVN * NH);
  float* aeb   = alloc((size_t)EVN * NH);
  float* amaxb = alloc((size_t)NN * NH);   // ordered-uint encoding
  float* asumb = alloc((size_t)NN * NH);
  float* Pab   = alloc((size_t)GN * DCH);
  float* Psg   = alloc((size_t)GN * DCH);
  float* denom = alloc((size_t)NN);
  // f16 region
  _Float16* H16 = (_Float16*)(WS + off);
  size_t hoff = 0;
  auto halloc = [&](size_t n) { _Float16* p = H16 + hoff; hoff += n; return p; };
  _Float16* w_q16   = halloc((size_t)8 * HD * DCH);   // per (layer,view): 512 x 128
  _Float16* w_k16   = halloc((size_t)8 * HD * DCH);
  _Float16* w_v16   = halloc((size_t)8 * HD * DCH);
  _Float16* w_sk16  = halloc((size_t)8 * DCH * DCH);  // 128 x 128
  _Float16* w_e1    = halloc((size_t)DCH * 64);       // enc_W1^T : 128 x 64
  _Float16* w_e2    = halloc((size_t)DCH * DCH);
  _Float16* w_pf    = halloc((size_t)DCH * 256);      // 128 x 256
  _Float16* w_m1    = halloc((size_t)DCH * 32);       // 128 x 32 (K=16 padded)
  _Float16* w_m2    = halloc((size_t)DCH * DCH);
  _Float16* w_t1    = halloc((size_t)DCH * 32);       // 128 x 32 (K=8 padded)
  _Float16* w_t2    = halloc((size_t)DCH * DCH);
  _Float16* w_h1    = halloc((size_t)DCH * 384);      // 128 x 384
  _Float16* x16     = halloc((size_t)NN * 64);
  _Float16* mkt16   = halloc((size_t)NN * 32);
  _Float16* trd16   = halloc((size_t)NN * 32);
  _Float16* h16     = halloc((size_t)NN * DCH);
  _Float16* act16a  = halloc((size_t)NN * DCH);
  _Float16* act16b  = halloc((size_t)NN * DCH);
  _Float16* pfcat16 = halloc((size_t)NN * 256);
  _Float16* z16     = halloc((size_t)NN * 384);

  const dim3 blk(256);
  auto gemm = [&](const _Float16* A, const _Float16* BT, const float* bias, float* C,
                  _Float16* C16, int M, int Nc, int Kp, int relu) {
    const int waves = ((M + 31) / 32) * (Nc / 32);
    gemm_wmma_kernel<<<dim3((waves + 7) / 8), blk, 0, stream>>>(A, BT, bias, C, C16, M, Nc, Kp, relu);
  };
  auto fill = [&](float* p, float v, int n) {
    fill_kernel<<<dim3((n + 255) / 256), blk, 0, stream>>>(p, v, n);
  };
  auto cvtT = [&](const float* B, _Float16* BT, int K, int Nc, int Kp) {
    cvt_transpose_kernel<<<dim3((Nc * Kp + 255) / 256), blk, 0, stream>>>(B, BT, K, Nc, Kp);
  };
  auto cvtP = [&](const float* A, _Float16* O, int rows, int K, int Kp) {
    cvt_pad_kernel<<<dim3((rows * Kp + 255) / 256), blk, 0, stream>>>(A, O, rows, K, Kp);
  };
  const int nodeBlocks = (NN + 7) / 8;
  const int edgeBlocks = (EVN * NH + 7) / 8;
  const int lineBlocks = (LLN + 7) / 8;

  // -------- one-time weight transpose+convert and input converts --------
  for (int po = 0; po < 8; ++po) {
    cvtT(Wq + (size_t)po * DCH * HD, w_q16 + (size_t)po * HD * DCH, DCH, HD, DCH);
    cvtT(Wk + (size_t)po * DCH * HD, w_k16 + (size_t)po * HD * DCH, DCH, HD, DCH);
    cvtT(Wv + (size_t)po * DCH * HD, w_v16 + (size_t)po * HD * DCH, DCH, HD, DCH);
    cvtT(Wskip + (size_t)po * DCH * DCH, w_sk16 + (size_t)po * DCH * DCH, DCH, DCH, DCH);
  }
  cvtT(eW1, w_e1, XDIM, DCH, 64);
  cvtT(eW2, w_e2, DCH, DCH, DCH);
  cvtT(pfW, w_pf, 256, DCH, 256);
  cvtT(mW1, w_m1, 16, DCH, 32);
  cvtT(mW2, w_m2, DCH, DCH, DCH);
  cvtT(tW1, w_t1, 8, DCH, 32);
  cvtT(tW2, w_t2, DCH, DCH, DCH);
  cvtT(hW1, w_h1, 384, DCH, 384);
  cvtP(x, x16, NN, XDIM, 64);
  cvtP(mkt, mkt16, NN, 16, 32);
  cvtP(trade, trd16, NN, 8, 32);

  // -------- encoder: h = LN(relu(x@W1+b1)@W2+b2) --------
  gemm(x16, w_e1, eb1, bufQ, act16a, NN, DCH, 64, 1);
  gemm(act16a, w_e2, eb2, bufK, nullptr, NN, DCH, DCH, 0);
  ln_kernel<<<nodeBlocks, blk, 0, stream>>>(bufK, nullptr, ln0g, ln0b, h, h16, NN);

  // -------- 2 layers x 4 views of TransformerConv --------
  for (int layer = 0; layer < 2; ++layer) {
    fill(fused, 0.f, NN * DCH);
    const float* gv  = layer ? g2 : g1;
    const float* lng = layer ? ln2g : ln1g;
    const float* lnb = layer ? ln2b : ln1b;
    for (int vi = 0; vi < NVW; ++vi) {
      const size_t po = (size_t)layer * NVW + vi;
      gemm(h16, w_q16 + po * HD * DCH, bq + po * HD, bufQ, nullptr, NN, HD, DCH, 0);
      gemm(h16, w_k16 + po * HD * DCH, bk + po * HD, bufK, nullptr, NN, HD, DCH, 0);
      gemm(h16, w_v16 + po * HD * DCH, bv + po * HD, bufV, nullptr, NN, HD, DCH, 0);
      gemm(h16, w_sk16 + po * DCH * DCH, bskip + po * DCH, skipb, nullptr, NN, DCH, DCH, 0);
      fill(bufO, 0.f, NN * HD);
      fill(amaxb, 0.f, NN * NH);   // 0u == -inf in ordered encoding
      fill(asumb, 0.f, NN * NH);
      const int*   sv  = src + (size_t)vi * EVN;
      const int*   dv  = dst + (size_t)vi * EVN;
      const float* ewv = ew + (size_t)vi * EVN;
      const float* Wev = We + po * HD;
      const float* Wbv = Wbeta + po * 3 * DCH;
      edge_alpha_kernel<<<edgeBlocks, blk, 0, stream>>>(bufQ, bufK, ewv, Wev, sv, dv,
                                                        alphb, (unsigned*)amaxb);
      edge_exp_kernel<<<(EVN * NH + 255) / 256, blk, 0, stream>>>(alphb, (unsigned*)amaxb,
                                                                  dv, aeb, asumb);
      edge_agg_kernel<<<edgeBlocks, blk, 0, stream>>>(bufV, ewv, Wev, sv, dv, aeb, asumb, bufO);
      combine_kernel<<<nodeBlocks, blk, 0, stream>>>(bufO, skipb, Wbv, gv, vi,
                                                     (vi >= 2) ? corrg : nullptr, fused);
    }
    ln_kernel<<<nodeBlocks, blk, 0, stream>>>(h, fused, lng, lnb, h, h16, NN);
  }

  // -------- portfolio pathway --------
  float* Vab   = bufQ;
  float* Vsg   = bufQ + (size_t)NN * DCH;
  float* pfout = bufV;
  fill(Pab, 0.f, GN * DCH);
  fill(Psg, 0.f, GN * DCH);
  fill(Vab, 0.f, NN * DCH);
  fill(Vsg, 0.f, NN * DCH);
  fill(denom, 0.f, NN);
  pf_accum_kernel<<<lineBlocks, blk, 0, stream>>>(h, pnod, pw, pgid, Pab, Psg, denom);
  pf_scatter_kernel<<<lineBlocks, blk, 0, stream>>>(h, pnod, pw, pgid, Pab, Psg, Vab, Vsg);
  pf_finish_kernel<<<nodeBlocks, blk, 0, stream>>>(Vab, Vsg, denom, pfcat16);
  gemm(pfcat16, w_pf, pfb, pfout, nullptr, NN, DCH, 256, 0);
  add_gate_kernel<<<(NN * DCH + 255) / 256, blk, 0, stream>>>(h, pfout, pfg, NN * DCH);

  // -------- mkt / trade MLPs --------
  float* mm = bufO + (size_t)NN * DCH;
  float* tt = bufO + 3 * (size_t)NN * DCH;
  gemm(mkt16, w_m1, mb1, bufO, act16a, NN, DCH, 32, 1);
  gemm(act16a, w_m2, mb2, mm, nullptr, NN, DCH, DCH, 0);
  gemm(trd16, w_t1, tb1, bufO + 2 * (size_t)NN * DCH, act16b, NN, DCH, 32, 1);
  gemm(act16b, w_t2, tb2, tt, nullptr, NN, DCH, DCH, 0);

  // -------- head: z=[h,m,t] -> relu(z@W1+b1) -> @W2+b2 --------
  float* hh = bufQ;
  concat3_kernel<<<(NN * 3 * DCH + 255) / 256, blk, 0, stream>>>(h, mm, tt, z16);
  gemm(z16, w_h1, hb1, hh, nullptr, NN, DCH, 384, 1);
  head_gemv_kernel<<<nodeBlocks, blk, 0, stream>>>(hh, hW2, hb2, (float*)d_out);
}
